// MultiHeadAttention_44032004719072
// MI455X (gfx1250) — matrix operand order repaired and checked by reading, not compiled, not run on hardware
//
#include <hip/hip_runtime.h>
#include <hip/hip_bf16.h>
#include <cmath>

// MHA + residual + LayerNorm for MI455X (gfx1250, wave32, WMMA).
// - bf16 V_WMMA_F32_16X16X32_BF16 for all GEMMs (8x K-depth vs the only f32
//   WMMA shape 16x16x4), f32 accumulation.
// - flash-style streaming attention: the 512 MB score matrix never touches
//   HBM (round-tripping it at 23.3 TB/s would cost ~45us vs ~6.5us for all
//   other traffic combined).
// - V is produced PRE-TRANSPOSED (Vt[b][h][hd][s]) by its projection GEMM, so
//   both K and V WMMA B-fragments in attention are contiguous b128 loads:
//   no LDS transpose gather on the consume side at all. The GEMM epilogue
//   uses branch-free affine addressing (rowStride/colStride/bOff scalars) so
//   both layouts share one clean unrolled store chain.
// - global_prefetch (gfx1250) pulls the next key block's K/Vt lines while the
//   current block's WMMAs run.

#define B_   2
#define S_   2048
#define D_   1024
#define H_   16
#define HD_  64
#define LN_EPS 1e-5f

typedef __bf16 bf16_t;
typedef __attribute__((ext_vector_type(16))) __bf16 v16bf;
typedef __attribute__((ext_vector_type(8)))  __bf16 v8bf;
typedef __attribute__((ext_vector_type(4)))  __bf16 v4bf;
typedef __attribute__((ext_vector_type(8)))  float  v8f;
typedef __attribute__((ext_vector_type(4)))  float  v4f;

static __device__ inline v16bf cat8(v8bf a, v8bf b) {
  return __builtin_shufflevector(a, b, 0,1,2,3,4,5,6,7,8,9,10,11,12,13,14,15);
}

static __device__ inline v8f v8f_zero() {
  v8f z = {0.f,0.f,0.f,0.f,0.f,0.f,0.f,0.f};
  return z;
}

// A fragment (16x32, bf16): lane l holds row (l&15); lanes 0-15 carry k
// chunks {0..7, 16..23}, lanes 16-31 carry {8..15, 24..31}.
static __device__ inline v16bf load_frag_a(const bf16_t* base, int ldm, int lane) {
  const int hf = (lane >> 4) & 1;
  const bf16_t* rp = base + (size_t)(lane & 15) * ldm + hf * 8;
  v8bf c0 = *(const v8bf*)(rp);
  v8bf c1 = *(const v8bf*)(rp + 16);
  return cat8(c0, c1);
}

// B fragment (32x16, bf16) for the case where column n of B is contiguous in
// memory at base + n*ldn. Lane l holds column (l&15); lanes 0-15 carry k
// chunks {0..7, 16..23}, lanes 16-31 carry {8..15, 24..31}.
static __device__ inline v16bf load_frag_bt(const bf16_t* base, size_t ldn, int lane) {
  const int hf = (lane >> 4) & 1;
  const bf16_t* rp = base + (size_t)(lane & 15) * ldn + hf * 8;
  v8bf c0 = *(const v8bf*)(rp);
  v8bf c1 = *(const v8bf*)(rp + 16);
  return cat8(c0, c1);
}

static __device__ inline v8f wmma_bf16(v16bf a, v16bf b, v8f c) {
  return __builtin_amdgcn_wmma_f32_16x16x32_bf16(
      /*neg_a=*/false, a, /*neg_b=*/false, b,
      /*c_mod=*/(short)0, c, /*reuse_a=*/false, /*reuse_b=*/false);
}

// ---------------------------------------------------------------- cvt f32->bf16
__global__ void __launch_bounds__(256) cvt_f32_bf16(
    const float* __restrict__ in, bf16_t* __restrict__ out, int n) {
  int i = (blockIdx.x * blockDim.x + threadIdx.x) * 4;
  if (i + 3 < n) {
    v4f x = *(const v4f*)(in + i);
    v4bf y;
    y[0] = (bf16_t)x[0]; y[1] = (bf16_t)x[1];
    y[2] = (bf16_t)x[2]; y[3] = (bf16_t)x[3];
    *(v4bf*)(out + i) = y;
  }
}

// ------------------------------------------------- Out = (X @ W^T + b) * scale
// One wave computes a 64x64 output tile (4x4 WMMA C tiles).
// Store address is affine: addr = mrow*rowStride + ncol*colStride + bb*bOff,
// with bb = mrow / S_ constant per tile (tiles never straddle the batch
// boundary). Row-major [M,N]: (N, 1, 0). Pre-transposed Vt[b][h][hd][s]:
// (1, S_, S_*D_ - S_)  [since (h*64+hd)*S == ncol*S and s == mrow - bb*S].
__global__ void __launch_bounds__(32) gemm_xwT(
    const bf16_t* __restrict__ X, const bf16_t* __restrict__ W,
    const float* __restrict__ bias, float outScale,
    size_t rowStride, size_t colStride, size_t bOff,
    bf16_t* __restrict__ Out, int M, int N, int K) {
  const int lane = threadIdx.x;
  const int n0 = blockIdx.x * 64;
  const int m0 = blockIdx.y * 64;

  v8f acc[4][4];
  #pragma unroll
  for (int mi = 0; mi < 4; ++mi)
    #pragma unroll
    for (int ni = 0; ni < 4; ++ni) acc[mi][ni] = v8f_zero();

  for (int k = 0; k < K; k += 32) {
    v16bf a[4], b[4];
    #pragma unroll
    for (int mi = 0; mi < 4; ++mi)
      a[mi] = load_frag_a(X + (size_t)(m0 + mi * 16) * K + k, K, lane);
    #pragma unroll
    for (int ni = 0; ni < 4; ++ni)
      b[ni] = load_frag_bt(W + (size_t)(n0 + ni * 16) * K + k, K, lane);
    #pragma unroll
    for (int mi = 0; mi < 4; ++mi)
      #pragma unroll
      for (int ni = 0; ni < 4; ++ni)
        acc[mi][ni] = wmma_bf16(a[mi], b[ni], acc[mi][ni]);
  }

  const int hf = lane >> 4, col = lane & 15;
  const size_t bbase = (size_t)(m0 >> 11) * bOff;  // m0 / S_ * bOff, per-tile
  #pragma unroll
  for (int ni = 0; ni < 4; ++ni) {
    const int ncol = n0 + ni * 16 + col;
    const float bv = bias[ncol];
    const size_t colOff = bbase + (size_t)ncol * colStride;
    #pragma unroll
    for (int mi = 0; mi < 4; ++mi) {
      size_t r = (size_t)(m0 + mi * 16 + 8 * hf) * rowStride + colOff;
      #pragma unroll
      for (int g = 0; g < 8; ++g) {
        Out[r] = (bf16_t)((acc[mi][ni][g] + bv) * outScale);
        r += rowStride;
      }
    }
  }
}

// ----------------------------------------------------- streaming attention
// One wave handles 32 query rows (2 m-tiles) of one (b,h): 16 WMMAs per
// 32-key block fed by 16 contiguous b128 fragment loads (K rows and Vt rows).
// Online softmax over ALL keys (reference softmaxes the full row); the
// post-softmax triu mask (keep k >= q) is applied only to the P feeding P@V.
// LDS is used solely to relayout P from C-fragment to A-fragment form.
// Output Z stored [b,h,s,hd] f32 contiguous == the reference's raw
// .view(b,s,-1) flat layout.
__global__ void __launch_bounds__(32) attn_fa(
    const bf16_t* __restrict__ Q, const bf16_t* __restrict__ Km,
    const bf16_t* __restrict__ Vt, float* __restrict__ Z) {
  const int lane = threadIdx.x;
  const int q0 = blockIdx.x * 32;
  const int h  = blockIdx.y;
  const int b  = blockIdx.z;
  const int hf = lane >> 4, col = lane & 15;

  __shared__ __align__(16) bf16_t Pt[32][32];

  const bf16_t* Qb = Q  + (size_t)(b * S_ + q0) * D_ + h * HD_;
  const bf16_t* Kb = Km + (size_t)b * S_ * D_ + h * HD_;
  const bf16_t* Vb = Vt + (size_t)b * S_ * D_ + (size_t)h * HD_ * S_;  // [hd][s]

  v16bf aq[2][2];
  #pragma unroll
  for (int mi = 0; mi < 2; ++mi) {
    aq[mi][0] = load_frag_a(Qb + (size_t)mi * 16 * D_ + 0,  D_, lane);
    aq[mi][1] = load_frag_a(Qb + (size_t)mi * 16 * D_ + 32, D_, lane);
  }

  v8f o[2][4];
  float mrow[2][8], lrow[2][8];
  #pragma unroll
  for (int mi = 0; mi < 2; ++mi) {
    #pragma unroll
    for (int nt = 0; nt < 4; ++nt) o[mi][nt] = v8f_zero();
    #pragma unroll
    for (int g = 0; g < 8; ++g) { mrow[mi][g] = -1e30f; lrow[mi][g] = 0.f; }
  }

  for (int kb = 0; kb < S_; kb += 32) {
    // Prefetch the next key block while this block's WMMAs run: 32 lanes
    // cover the next 32 K rows, and (spread 2 apart) the 64 Vt dim-rows.
    if (kb + 32 < S_) {
      __builtin_prefetch(Kb + (size_t)(kb + 32 + lane) * D_, 0, 1);
      __builtin_prefetch(Vb + (size_t)(lane * 2) * S_ + kb + 32, 0, 1);
    }

    // scores S = Qs @ K^T : 2 m-tiles x 2 key-tiles, K-dim 64 -> 8 WMMAs
    v16bf bk[2][2];
    #pragma unroll
    for (int nt = 0; nt < 2; ++nt) {
      bk[nt][0] = load_frag_bt(Kb + (size_t)(kb + nt * 16) * D_ + 0,  D_, lane);
      bk[nt][1] = load_frag_bt(Kb + (size_t)(kb + nt * 16) * D_ + 32, D_, lane);
    }
    v8f sc[2][2];
    #pragma unroll
    for (int mi = 0; mi < 2; ++mi)
      #pragma unroll
      for (int nt = 0; nt < 2; ++nt) {
        v8f s = v8f_zero();
        s = wmma_bf16(aq[mi][0], bk[nt][0], s);
        s = wmma_bf16(aq[mi][1], bk[nt][1], s);
        sc[mi][nt] = s;
      }

    // online softmax (full-row denominator), then post-softmax mask -> P
    #pragma unroll
    for (int mi = 0; mi < 2; ++mi) {
      #pragma unroll
      for (int g = 0; g < 8; ++g) {
        float x0 = sc[mi][0][g], x1 = sc[mi][1][g];
        float mx = fmaxf(x0, x1);
        #pragma unroll
        for (int msk = 1; msk < 16; msk <<= 1)
          mx = fmaxf(mx, __shfl_xor(mx, msk, 32));
        const float mnew = fmaxf(mrow[mi][g], mx);
        const float rs = __expf(mrow[mi][g] - mnew);
        float p0 = __expf(x0 - mnew);
        float p1 = __expf(x1 - mnew);
        float sm = p0 + p1;
        #pragma unroll
        for (int msk = 1; msk < 16; msk <<= 1)
          sm += __shfl_xor(sm, msk, 32);
        lrow[mi][g] = lrow[mi][g] * rs + sm;
        mrow[mi][g] = mnew;
        o[mi][0][g] *= rs; o[mi][1][g] *= rs;
        o[mi][2][g] *= rs; o[mi][3][g] *= rs;
        const int qrow = q0 + mi * 16 + g + 8 * hf;
        float p0m = (kb + col >= qrow) ? p0 : 0.f;       // triu: keep k >= q
        float p1m = (kb + 16 + col >= qrow) ? p1 : 0.f;
        Pt[mi * 16 + g + 8 * hf][col]      = (bf16_t)p0m;
        Pt[mi * 16 + g + 8 * hf][16 + col] = (bf16_t)p1m;
      }
    }
    __syncthreads();

    // P (C-layout in LDS) -> A fragments (one per m-tile)
    v16bf ap[2];
    #pragma unroll
    for (int mi = 0; mi < 2; ++mi) {
      const bf16_t* pr = &Pt[mi * 16 + col][hf * 8];
      ap[mi] = cat8(*(const v8bf*)(pr), *(const v8bf*)(pr + 16));
    }

    // P @ V : V B-fragments are contiguous rows of Vt -> plain b128 loads
    #pragma unroll
    for (int nt = 0; nt < 4; ++nt) {
      v16bf bv = load_frag_bt(Vb + (size_t)(nt * 16) * S_ + kb, S_, lane);
      o[0][nt] = wmma_bf16(ap[0], bv, o[0][nt]);
      o[1][nt] = wmma_bf16(ap[1], bv, o[1][nt]);
    }
    __syncthreads();
  }

  float* Zb = Z + (((size_t)b * H_ + h) * S_ + q0) * HD_;
  #pragma unroll
  for (int mi = 0; mi < 2; ++mi) {
    float inv[8];
    #pragma unroll
    for (int g = 0; g < 8; ++g) inv[g] = 1.0f / lrow[mi][g];
    #pragma unroll
    for (int nt = 0; nt < 4; ++nt)
      #pragma unroll
      for (int g = 0; g < 8; ++g)
        Zb[(size_t)(mi * 16 + g + 8 * hf) * HD_ + nt * 16 + col] =
            o[mi][nt][g] * inv[g];
  }
}

// ------------------------------------------------ residual + LayerNorm
__global__ void __launch_bounds__(256) resid_ln(
    const float* __restrict__ emb, const float* __restrict__ Z,
    const float* __restrict__ gamma, const float* __restrict__ beta,
    float* __restrict__ out) {
  const int row = blockIdx.x;
  const int t = threadIdx.x;
  const float* x = emb + (size_t)row * D_;
  const float* z = Z   + (size_t)row * D_;
  __shared__ float red[256];

  float v[4];
  float s = 0.f;
  #pragma unroll
  for (int i = 0; i < 4; ++i) {
    const int idx = t + i * 256;
    v[i] = x[idx] + z[idx];
    s += v[i];
  }
  red[t] = s; __syncthreads();
  for (int w = 128; w > 0; w >>= 1) {
    if (t < w) red[t] += red[t + w];
    __syncthreads();
  }
  const float mu = red[0] * (1.0f / D_);
  __syncthreads();

  float vs = 0.f;
  #pragma unroll
  for (int i = 0; i < 4; ++i) {
    const float d = v[i] - mu;
    vs += d * d;
  }
  red[t] = vs; __syncthreads();
  for (int w = 128; w > 0; w >>= 1) {
    if (t < w) red[t] += red[t + w];
    __syncthreads();
  }
  const float rstd = rsqrtf(red[0] * (1.0f / D_) + LN_EPS);

  #pragma unroll
  for (int i = 0; i < 4; ++i) {
    const int idx = t + i * 256;
    out[(size_t)row * D_ + idx] = (v[i] - mu) * rstd * gamma[idx] + beta[idx];
  }
}

// ---------------------------------------------------------------- launcher
extern "C" void kernel_launch(void* const* d_in, const int* in_sizes, int n_in,
                              void* d_out, int out_size, void* d_ws, size_t ws_size,
                              hipStream_t stream) {
  const float* emb   = (const float*)d_in[0];
  const float* Wq    = (const float*)d_in[1];
  const float* bq    = (const float*)d_in[2];
  const float* Wk    = (const float*)d_in[3];
  const float* bk    = (const float*)d_in[4];
  const float* Wv    = (const float*)d_in[5];
  const float* bv    = (const float*)d_in[6];
  const float* gamma = (const float*)d_in[7];
  const float* beta  = (const float*)d_in[8];
  float* out = (float*)d_out;

  const size_t MD = (size_t)B_ * S_ * D_;   // 4,194,304
  const size_t DD = (size_t)D_ * D_;        // 1,048,576

  char* ws = (char*)d_ws;
  size_t off = 0;
  auto carve = [&](size_t bytes) -> void* {
    void* p = ws + off;
    off += (bytes + 255) & ~size_t(255);
    return p;
  };
  bf16_t* Xbf  = (bf16_t*)carve(MD * 2);
  bf16_t* Wqbf = (bf16_t*)carve(DD * 2);
  bf16_t* Wkbf = (bf16_t*)carve(DD * 2);
  bf16_t* Wvbf = (bf16_t*)carve(DD * 2);
  bf16_t* Qbf  = (bf16_t*)carve(MD * 2);
  bf16_t* Kbf  = (bf16_t*)carve(MD * 2);
  bf16_t* Vtbf = (bf16_t*)carve(MD * 2);
  float*  Zf   = (float*)carve(MD * 4);

  cvt_f32_bf16<<<(int)((MD / 4 + 255) / 256), 256, 0, stream>>>(emb, Xbf, (int)MD);
  cvt_f32_bf16<<<(int)((DD / 4 + 255) / 256), 256, 0, stream>>>(Wq, Wqbf, (int)DD);
  cvt_f32_bf16<<<(int)((DD / 4 + 255) / 256), 256, 0, stream>>>(Wk, Wkbf, (int)DD);
  cvt_f32_bf16<<<(int)((DD / 4 + 255) / 256), 256, 0, stream>>>(Wv, Wvbf, (int)DD);

  const int M = B_ * S_;
  dim3 gg(D_ / 64, M / 64);
  const float qscale = 1.0f / sqrtf((float)S_);  // reference scales by sqrt(seq_len)
  // Q, K: row-major [M, D].  V: pre-transposed Vt[b][h][hd][s].
  gemm_xwT<<<gg, 32, 0, stream>>>(Xbf, Wqbf, bq, qscale,
                                  (size_t)D_, (size_t)1, (size_t)0,
                                  Qbf, M, D_, D_);
  gemm_xwT<<<gg, 32, 0, stream>>>(Xbf, Wkbf, bk, 1.0f,
                                  (size_t)D_, (size_t)1, (size_t)0,
                                  Kbf, M, D_, D_);
  gemm_xwT<<<gg, 32, 0, stream>>>(Xbf, Wvbf, bv, 1.0f,
                                  (size_t)1, (size_t)S_,
                                  (size_t)S_ * D_ - (size_t)S_,
                                  Vtbf, M, D_, D_);

  dim3 ga(S_ / 32, H_, B_);
  attn_fa<<<ga, 32, 0, stream>>>(Qbf, Kbf, Vtbf, Zf);

  resid_ln<<<M, 256, 0, stream>>>(emb, Zf, gamma, beta, out);
}
